// LSTMTagger_6983616823935
// MI455X (gfx1250) — matrix operand order repaired and checked by reading, not compiled, not run on hardware
//
#include <hip/hip_runtime.h>
#include <hip/hip_bf16.h>

// ---------------------------------------------------------------------------
// LSTM tagger for MI455X (gfx1250, wave32, WMMA, TDM).
//   S=8192 words, L=16 chars, WORD_DIM=256, CHAR_DIM=64,
//   CHAR_HID=128, WORD_HID=512, N_TAG=64.
//
// Roofline reasoning:
//  * char LSTM + input GEMM + tag head: batched bf16 WMMA GEMMs, everything
//    L2-resident (~85 MB workspace << 192 MB L2); HBM never the limiter.
//  * word recurrence is the true bottleneck (8192 serial steps). W_hh (2 MB
//    bf16) re-streamed from L2 each step by one WGP would cost ~16 GB of L2
//    traffic through one WGP. Instead: 16 WGs, each owns 32 h-cols -> 128
//    gate rows -> a 128 KB weight slice resident in LDS (loaded ONCE via the
//    Tensor Data Mover), per-step cross-WG sync via release/acquire flag
//    chain on a double-buffered h vector. Inner loop = LDS-fed WMMA only.
// ---------------------------------------------------------------------------

typedef __attribute__((ext_vector_type(16))) __bf16 v16bf;
typedef __attribute__((ext_vector_type(8)))  float  v8f;
typedef __attribute__((ext_vector_type(4)))  unsigned v4u;
typedef __attribute__((ext_vector_type(8)))  int      v8i;
typedef __attribute__((ext_vector_type(4)))  int      v4i;

union BF16x16 { v16bf v; unsigned u[8]; unsigned short h[16]; };

__device__ __forceinline__ unsigned short f2bf_bits(float f) {
    union { float f; unsigned u; } cv; cv.f = f;
    unsigned r = cv.u + 0x7FFFu + ((cv.u >> 16) & 1u);   // round-to-nearest-even
    return (unsigned short)(r >> 16);
}

__device__ __forceinline__ float sigf(float x) { return 1.0f / (1.0f + __expf(-x)); }

// B fragment: lane (n = colbase + lane%16) holds 16 contiguous K values,
// kbase = kb*32 + 16*(lane/16). W stored row-major [N][K] bf16 (== B^T).
__device__ __forceinline__ v16bf load_bfrag(const unsigned short* W, int ldK,
                                            int n, int kbase) {
    return *(const v16bf*)(W + (size_t)n * ldK + kbase);
}

// A fragment (16-bit A 16x32): lane row m = lane%16, hf = lane/16.
// Elements 0..7 -> K = kb*32 + 16*hf + e ; elements 8..15 -> +8.
__device__ __forceinline__ v16bf load_afrag(const unsigned short* A, int ldK,
                                            int row, int kb, int hf) {
    BF16x16 r;
    const unsigned* p0 = (const unsigned*)(A + (size_t)row * ldK + kb * 32 + 16 * hf);
    const unsigned* p1 = (const unsigned*)(A + (size_t)row * ldK + kb * 32 + 8 + 16 * hf);
#pragma unroll
    for (int i = 0; i < 4; ++i) { r.u[i] = p0[i]; r.u[4 + i] = p1[i]; }
    return r.v;
}

__device__ __forceinline__ v8f wmma_bf16(v16bf a, v16bf b, v8f c) {
    return __builtin_amdgcn_wmma_f32_16x16x32_bf16(false, a, false, b,
                                                   (short)0, c, false, false);
}

// ---------------------------------------------------------------------------
// Tensor Data Mover: 2-D tile load (rows x row_elems bf16) global -> LDS.
// Guarded: 5-arg builtin on clang-22 (ROCm 7.2), 6-arg on clang>=23,
// cooperative-copy fallback if the builtin is absent.
// ---------------------------------------------------------------------------
#if __has_builtin(__builtin_amdgcn_tensor_load_to_lds)
#define HAVE_TDM 1
__device__ __forceinline__ void tdm_load_2d(unsigned lds_byte_off, const void* gptr,
                                            unsigned rows, unsigned row_elems,
                                            unsigned row_stride_elems) {
    unsigned long long ga = (unsigned long long)(size_t)gptr;
    v4u g0 = {}; v8i g1 = {}; v4i g2 = {}; v4i g3 = {};
    g0[0] = 1u;                                           // D# count = 1 (user)
    g0[1] = lds_byte_off;                                 // lds_addr (bytes)
    g0[2] = (unsigned)(ga & 0xFFFFFFFFu);                 // global_addr[31:0]
    g0[3] = (unsigned)((ga >> 32) & 0x01FFFFFFu) | (2u << 30); // [56:32] | type=2
    g1[0] = (int)(1u << 16);                              // data_size=1 (2B)
    g1[1] = (int)((row_elems & 0xFFFFu) << 16);           // tensor_dim0[15:0]
    g1[2] = (int)(((row_elems >> 16) & 0xFFFFu) | ((rows & 0xFFFFu) << 16));
    g1[3] = (int)(((rows >> 16) & 0xFFFFu) | ((row_elems & 0xFFFFu) << 16)); // tile_dim0
    g1[4] = (int)(rows & 0xFFFFu);                        // tile_dim1 (tile_dim2=0)
    g1[5] = (int)row_stride_elems;                        // tensor_dim0_stride[31:0]
    g1[6] = 0; g1[7] = 0;
#if defined(__clang_major__) && (__clang_major__ >= 23)
    v8i g4 = {};
    __builtin_amdgcn_tensor_load_to_lds(g0, g1, g2, g3, g4, 0);
#else
    __builtin_amdgcn_tensor_load_to_lds(g0, g1, g2, g3, 0);
#endif
}
#endif

// ---------------------------------------------------------------------------
// Prep kernels
// ---------------------------------------------------------------------------
__global__ void k_cvt(const float* __restrict__ s, unsigned short* __restrict__ d, int n) {
    int i = blockIdx.x * blockDim.x + threadIdx.x;
    if (i < n) d[i] = f2bf_bits(s[i]);
}

__global__ void k_bias(const float* __restrict__ a, const float* __restrict__ b,
                       float* __restrict__ o, int n) {
    int i = blockIdx.x * blockDim.x + threadIdx.x;
    if (i < n) o[i] = a[i] + b[i];
}

__global__ void k_gather(const int* __restrict__ x, const float* __restrict__ emb,
                         unsigned short* __restrict__ o) {
    int i = blockIdx.x * blockDim.x + threadIdx.x;
    int s = i >> 8, d = i & 255;
    o[i] = f2bf_bits(emb[(size_t)x[s] * 256 + d]);
}

// ---------------------------------------------------------------------------
// Char LSTM: 512 WGs x 16 words each, 256 thr (8 waves), 16 time steps.
// ---------------------------------------------------------------------------
__global__ __launch_bounds__(256) void k_char_lstm(
    const int* __restrict__ chars, const int* __restrict__ lengths,
    const float* __restrict__ char_emb,
    const unsigned short* __restrict__ Wih,   // [512][64]  bf16
    const unsigned short* __restrict__ Whh,   // [512][128] bf16
    const float* __restrict__ cb,             // [512] = bih + bhh
    unsigned short* __restrict__ cf_bf)       // [8192][128] bf16 cell-state out
{
    __shared__ unsigned short sH[16 * 128];
    __shared__ float          sC[16 * 128];
    __shared__ float          sG[16 * 512];

    const int tid = threadIdx.x, wave = tid >> 5, lane = tid & 31;
    const int hf = lane >> 4, m = lane & 15;
    const int sbase = blockIdx.x * 16;

    for (int i = tid; i < 16 * 128; i += 256) { sH[i] = 0; sC[i] = 0.0f; }
    __syncthreads();

    for (int t = 0; t < 16; ++t) {
        const int cid = chars[(sbase + m) * 16 + t];
        const float* er = char_emb + (size_t)cid * 64;
        v16bf ax[2];
#pragma unroll
        for (int kb = 0; kb < 2; ++kb) {
            BF16x16 r;
#pragma unroll
            for (int i = 0; i < 4; ++i) {
                int k0 = kb * 32 + 16 * hf + 2 * i;
                int k1 = k0 + 8;
                r.u[i]     = f2bf_bits(er[k0]) | ((unsigned)f2bf_bits(er[k0 + 1]) << 16);
                r.u[4 + i] = f2bf_bits(er[k1]) | ((unsigned)f2bf_bits(er[k1 + 1]) << 16);
            }
            ax[kb] = r.v;
        }
        v16bf ah[4];
#pragma unroll
        for (int kb = 0; kb < 4; ++kb) ah[kb] = load_afrag(sH, 128, m, kb, hf);

#pragma unroll
        for (int ct = 0; ct < 4; ++ct) {
            const int n = wave * 64 + ct * 16 + m;
            v8f acc = {};
#pragma unroll
            for (int kb = 0; kb < 2; ++kb)
                acc = wmma_bf16(ax[kb], load_bfrag(Wih, 64, n, kb * 32 + 16 * hf), acc);
#pragma unroll
            for (int kb = 0; kb < 4; ++kb)
                acc = wmma_bf16(ah[kb], load_bfrag(Whh, 128, n, kb * 32 + 16 * hf), acc);
#pragma unroll
            for (int r = 0; r < 8; ++r) sG[(r + 8 * hf) * 512 + n] = acc[r];
        }
        __syncthreads();

        for (int j = tid; j < 16 * 128; j += 256) {
            int row = j >> 7, col = j & 127;
            const float* gr = sG + row * 512;
            float iv = sigf (gr[col]       + cb[col]);
            float fv = sigf (gr[col + 128] + cb[col + 128]);
            float gv = tanhf(gr[col + 256] + cb[col + 256]);
            float ov = sigf (gr[col + 384] + cb[col + 384]);
            float cn = fv * sC[j] + iv * gv;
            float hn = ov * tanhf(cn);
            if (t < lengths[sbase + row]) { sC[j] = cn; sH[j] = f2bf_bits(hn); }
        }
        __syncthreads();
    }
    for (int j = tid; j < 16 * 128; j += 256)
        cf_bf[(size_t)sbase * 128 + j] = f2bf_bits(sC[j]);
}

// ---------------------------------------------------------------------------
// Word input GEMM: Gw[8192][2048] = [we|cf] @ w_Wih^T + bias  (fp32 out)
// ---------------------------------------------------------------------------
__global__ __launch_bounds__(256) void k_gemm_wih(
    const unsigned short* __restrict__ we_bf,  // [8192][256]
    const unsigned short* __restrict__ cf_bf,  // [8192][128]
    const unsigned short* __restrict__ Wih,    // [2048][384]
    const float* __restrict__ wb,              // [2048]
    float* __restrict__ Gw)                    // [8192][2048]
{
    const int tid = threadIdx.x, wave = tid >> 5, lane = tid & 31;
    const int hf = lane >> 4, m = lane & 15;
    const int wg = blockIdx.x;
    const int rb = (wg >> 4) * 16;
    const int n  = (wg & 15) * 128 + wave * 16 + m;
    const int row = rb + m;

    v8f acc = {};
#pragma unroll
    for (int kb = 0; kb < 12; ++kb) {
        v16bf a = (kb < 8) ? load_afrag(we_bf, 256, row, kb, hf)
                           : load_afrag(cf_bf, 128, row, kb - 8, hf);
        acc = wmma_bf16(a, load_bfrag(Wih, 384, n, kb * 32 + 16 * hf), acc);
    }
    const float bias = wb[n];
#pragma unroll
    for (int r = 0; r < 8; ++r)
        Gw[(size_t)(rb + r + 8 * hf) * 2048 + n] = acc[r] + bias;
}

// ---------------------------------------------------------------------------
// Word LSTM recurrence, sharded over NWG=16 chained workgroups.
// WG b owns h cols [32b,32b+32) -> gate rows {512g+32b+r : g<4, r<32}.
// Its 128x512 bf16 W_hh slice (128 KB) is loaded ONCE into LDS via TDM.
// Per step: LDS-fed WMMA matvec -> gate update -> publish 32 h values to
// a double-buffered global h, release-fence + flag; spin/acquire next step.
// ---------------------------------------------------------------------------
#define NWG 16

__global__ __launch_bounds__(256) void k_word_rec(
    const unsigned short* __restrict__ Whh,   // [2048][512] bf16
    const float* __restrict__ Gw,             // [8192][2048] (biases included)
    unsigned short* __restrict__ hglob,       // [2][512] bf16 double buffer
    int* __restrict__ cnt,                    // [8192] step flags (pre-zeroed)
    unsigned short* __restrict__ hs_bf)       // [8192][512] bf16 out
{
    __shared__ unsigned short sWhh[128 * 512];   // 128 KB weight slice
    __shared__ unsigned short sHb[512];          // current h (bf16)
    __shared__ float          sGt[128];          // this WG's gate slice
    __shared__ float          sCl[32];           // this WG's c slice

    const int tid = threadIdx.x, wave = tid >> 5, lane = tid & 31;
    const int hf = lane >> 4, m = lane & 15;
    const int b  = blockIdx.x;
    const int g  = wave >> 1;                    // gate block 0..3
    const int sub = (wave & 1) * 16;             // sub-tile inside block
    const int nloc = wave * 16 + m;              // row in LDS slice
    const int jg = 512 * g + 32 * b + sub + m;   // global gate row

    // ---- one-time weight-slice preload (4 blocks of 32 rows) ----
#ifdef HAVE_TDM
    if (tid < 32) {
        unsigned lds0 = (unsigned)(size_t)(const void*)&sWhh[0];
#pragma unroll
        for (int gg = 0; gg < 4; ++gg)
            tdm_load_2d(lds0 + (unsigned)gg * 32 * 512 * 2,
                        Whh + (size_t)(512 * gg + 32 * b) * 512,
                        32u, 512u, 512u);
        __builtin_amdgcn_s_wait_tensorcnt(0);
    }
#else
    for (int gg = 0; gg < 4; ++gg) {
        const unsigned* src = (const unsigned*)(Whh + (size_t)(512 * gg + 32 * b) * 512);
        unsigned* dst = (unsigned*)&sWhh[gg * 32 * 512];
        for (int i = tid; i < 32 * 512 / 2; i += 256) dst[i] = src[i];
    }
#endif
    if (tid < 512) sHb[tid] = 0;
    if (tid < 32)  sCl[tid] = 0.0f;
    __syncthreads();

    for (int t = 0; t < 8192; ++t) {
        if (t > 0) {
            if (tid == 0) {                      // acquire previous step
                volatile int* vc = cnt + (t - 1);
                while (*vc != NWG) __builtin_amdgcn_s_sleep(1);
            }
            __syncthreads();
            __threadfence();
            const unsigned* src = (const unsigned*)(hglob + ((t - 1) & 1) * 512);
            ((unsigned*)sHb)[tid] = src[tid];    // 256 u32 == 512 bf16
            __syncthreads();
        }

        // matvec: gates_slice = h @ Whh_slice^T  (h replicated across A rows)
        v8f accA = {}, accB = {};
#pragma unroll
        for (int kb = 0; kb < 16; kb += 2) {
            v16bf a0 = load_afrag(sHb, 512, 0, kb,     hf);
            v16bf a1 = load_afrag(sHb, 512, 0, kb + 1, hf);
            accA = wmma_bf16(a0, load_bfrag(sWhh, 512, nloc, kb * 32 + 16 * hf), accA);
            accB = wmma_bf16(a1, load_bfrag(sWhh, 512, nloc, (kb + 1) * 32 + 16 * hf), accB);
        }
        v8f acc = accA + accB;
        if (t + 1 < 8192 && lane < 16)
            __builtin_prefetch(&Gw[(size_t)(t + 1) * 2048 + jg], 0, 1);
        if (lane < 16)
            sGt[wave * 16 + lane] = acc[0] + Gw[(size_t)t * 2048 + jg];
        __syncthreads();

        // gate update for this WG's 32 h columns (gate order i,f,g,o)
        if (tid < 32) {
            float iv = sigf (sGt[tid]);
            float fv = sigf (sGt[32 + tid]);
            float gv = tanhf(sGt[64 + tid]);
            float ov = sigf (sGt[96 + tid]);
            float cn = fv * sCl[tid] + iv * gv;
            sCl[tid] = cn;
            unsigned short hb = f2bf_bits(ov * tanhf(cn));
            hglob[(t & 1) * 512 + 32 * b + tid] = hb;
            hs_bf[(size_t)t * 512 + 32 * b + tid] = hb;
        }
        __threadfence();                         // release our h slice
        __syncthreads();
        if (tid == 0) atomicAdd(cnt + t, 1);
    }
}

// ---------------------------------------------------------------------------
// Tag head: logits = hs @ lin_W^T + b, then row log_softmax.
// ---------------------------------------------------------------------------
__global__ __launch_bounds__(128) void k_tagger(
    const unsigned short* __restrict__ hs_bf,  // [8192][512]
    const unsigned short* __restrict__ linW,   // [64][512]
    const float* __restrict__ lin_b,           // [64]
    float* __restrict__ out)                   // [8192][64]
{
    __shared__ float sL[16 * 64];
    const int tid = threadIdx.x, wave = tid >> 5, lane = tid & 31;
    const int hf = lane >> 4, m = lane & 15;
    const int rb = blockIdx.x * 16;
    const int n  = wave * 16 + m;

    v8f acc = {};
#pragma unroll
    for (int kb = 0; kb < 16; ++kb)
        acc = wmma_bf16(load_afrag(hs_bf, 512, rb + m, kb, hf),
                        load_bfrag(linW, 512, n, kb * 32 + 16 * hf), acc);
    const float bias = lin_b[n];
#pragma unroll
    for (int r = 0; r < 8; ++r) sL[(r + 8 * hf) * 64 + n] = acc[r] + bias;
    __syncthreads();

    if (tid < 16) {
        const float* rowp = sL + tid * 64;
        float mx = -1e30f;
        for (int j = 0; j < 64; ++j) mx = fmaxf(mx, rowp[j]);
        float s = 0.0f;
        for (int j = 0; j < 64; ++j) s += __expf(rowp[j] - mx);
        float l = __logf(s);
        float* op = out + (size_t)(rb + tid) * 64;
        for (int j = 0; j < 64; ++j) op[j] = rowp[j] - mx - l;
    }
}

// ---------------------------------------------------------------------------
extern "C" void kernel_launch(void* const* d_in, const int* in_sizes, int n_in,
                              void* d_out, int out_size, void* d_ws, size_t ws_size,
                              hipStream_t stream) {
    (void)in_sizes; (void)n_in; (void)out_size; (void)ws_size;

    const int*   x        = (const int*)  d_in[0];
    const int*   chars    = (const int*)  d_in[1];
    const int*   lengths  = (const int*)  d_in[2];
    const float* word_emb = (const float*)d_in[3];
    const float* char_emb = (const float*)d_in[4];
    const float* c_Wih    = (const float*)d_in[5];
    const float* c_Whh    = (const float*)d_in[6];
    const float* c_bih    = (const float*)d_in[7];
    const float* c_bhh    = (const float*)d_in[8];
    const float* w_Wih    = (const float*)d_in[9];
    const float* w_Whh    = (const float*)d_in[10];
    const float* w_bih    = (const float*)d_in[11];
    const float* w_bhh    = (const float*)d_in[12];
    const float* lin_W    = (const float*)d_in[13];
    const float* lin_b    = (const float*)d_in[14];
    float* out = (float*)d_out;

    char* base = (char*)d_ws;
    size_t off = 0;
    auto take = [&](size_t bytes) -> char* {
        char* p = base + off;
        off += (bytes + 255) & ~(size_t)255;
        return p;
    };
    unsigned short* cWih_bf = (unsigned short*)take(512 * 64   * 2);
    unsigned short* cWhh_bf = (unsigned short*)take(512 * 128  * 2);
    unsigned short* wWih_bf = (unsigned short*)take(2048 * 384 * 2);
    unsigned short* wWhh_bf = (unsigned short*)take(2048 * 512 * 2);
    unsigned short* linW_bf = (unsigned short*)take(64 * 512   * 2);
    unsigned short* we_bf   = (unsigned short*)take((size_t)8192 * 256 * 2);
    unsigned short* cf_bf   = (unsigned short*)take((size_t)8192 * 128 * 2);
    unsigned short* hs_bf   = (unsigned short*)take((size_t)8192 * 512 * 2);
    unsigned short* hglob   = (unsigned short*)take(2 * 512 * 2);
    int*            cnt     = (int*)take(8192 * 4);
    float*          cb      = (float*)take(512 * 4);
    float*          wb      = (float*)take(2048 * 4);
    float*          Gw      = (float*)take((size_t)8192 * 2048 * 4);

    auto cvt = [&](const float* s, unsigned short* d, int n) {
        k_cvt<<<(n + 255) / 256, 256, 0, stream>>>(s, d, n);
    };
    cvt(c_Wih, cWih_bf, 512 * 64);
    cvt(c_Whh, cWhh_bf, 512 * 128);
    cvt(w_Wih, wWih_bf, 2048 * 384);
    cvt(w_Whh, wWhh_bf, 2048 * 512);
    cvt(lin_W, linW_bf, 64 * 512);
    k_bias<<<2, 256, 0, stream>>>(c_bih, c_bhh, cb, 512);
    k_bias<<<8, 256, 0, stream>>>(w_bih, w_bhh, wb, 2048);
    k_gather<<<(8192 * 256) / 256, 256, 0, stream>>>(x, word_emb, we_bf);
    hipMemsetAsync(cnt, 0, 8192 * sizeof(int), stream);   // step flags

    k_char_lstm<<<512, 256, 0, stream>>>(chars, lengths, char_emb,
                                         cWih_bf, cWhh_bf, cb, cf_bf);
    k_gemm_wih<<<8192, 256, 0, stream>>>(we_bf, cf_bf, wWih_bf, wb, Gw);
    k_word_rec<<<NWG, 256, 0, stream>>>(wWhh_bf, Gw, hglob, cnt, hs_bf);
    k_tagger<<<512, 128, 0, stream>>>(hs_bf, linW_bf, lin_b, out);
}
